// CompatibleLearningLoss_2353642078403
// MI455X (gfx1250) — matrix operand order repaired and checked by reading, not compiled, not run on hardware
//
#include <hip/hip_runtime.h>
#include <hip/hip_bf16.h>
#include <math.h>

// Problem constants (match reference)
#define Nn 128
#define Dn 512
#define Cn 8192
#define Qn 32768
#define NB (Qn / 16)   // 2048 column tiles of 16 queue rows

typedef __attribute__((ext_vector_type(16))) __bf16 v16bf;
typedef __attribute__((ext_vector_type(8)))  float  v8f;

// ---------------------------------------------------------------------------
// Shared helper: load one 32xK B-slab lane-chunk (16 f32) and convert to the
// bf16 B operand in this file's contiguous k order, same as A (lanes 0-15: K 0..15, lanes 16-31: K 16..31).
// ---------------------------------------------------------------------------
__device__ inline v16bf load_b16(const float* __restrict__ bp)
{
    float4 f0 = *reinterpret_cast<const float4*>(bp);
    float4 f1 = *reinterpret_cast<const float4*>(bp + 4);
    float4 f2 = *reinterpret_cast<const float4*>(bp + 8);
    float4 f3 = *reinterpret_cast<const float4*>(bp + 12);
    union { v16bf v; __bf16 h[16]; } b;
    b.h[0]  = (__bf16)f0.x; b.h[1]  = (__bf16)f0.y; b.h[2]  = (__bf16)f0.z; b.h[3]  = (__bf16)f0.w;
    b.h[4]  = (__bf16)f1.x; b.h[5]  = (__bf16)f1.y; b.h[6]  = (__bf16)f1.z; b.h[7]  = (__bf16)f1.w;
    b.h[8]  = (__bf16)f2.x; b.h[9]  = (__bf16)f2.y; b.h[10] = (__bf16)f2.z; b.h[11] = (__bf16)f2.w;
    b.h[12] = (__bf16)f3.x; b.h[13] = (__bf16)f3.y; b.h[14] = (__bf16)f3.z; b.h[15] = (__bf16)f3.w;
    return b.v;
}

__device__ inline v16bf load_a16(const unsigned short* __restrict__ ap)
{
    union { v16bf v; uint4 q[2]; } a;
    a.q[0] = *reinterpret_cast<const uint4*>(ap);
    a.q[1] = *reinterpret_cast<const uint4*>(ap + 8);
    return a.v;
}

// ---------------------------------------------------------------------------
// Single-A GEMM micro-tile: D(16x16,f32) += A(16xK,bf16) * B(Kx16, f32->bf16).
// ---------------------------------------------------------------------------
template <int K>
__device__ inline v8f gemm_tile(const unsigned short* __restrict__ Abf, int lda,
                                int arow, const float* __restrict__ brow, int lane)
{
    v8f c = {};
    const int aoff = (lane & 16) ? 16 : 0;   // A: K 0..15 vs 16..31
    const int koff = (lane & 16) ? 16 : 0;   // B: K 0..15 vs 16..31
    const unsigned short* ap = Abf + (size_t)arow * lda + aoff;
#pragma unroll 4
    for (int kb = 0; kb < K; kb += 32) {
        const float* bp = brow + kb + koff;
        __builtin_prefetch(bp + 128, 0, 1);            // global_prefetch ahead of stream
        v16bf a = load_a16(ap + kb);
        v16bf b = load_b16(bp);
        c = __builtin_amdgcn_wmma_f32_16x16x32_bf16(false, a, false, b,
                                                    (short)0, c, false, false);
    }
    return c;
}

// ---------------------------------------------------------------------------
// Dual-A GEMM micro-tile: both feat-path GEMMs share B (feat_queue rows), so
// load + convert B once per k-step and issue two independent WMMAs.
// ---------------------------------------------------------------------------
template <int K>
__device__ inline void gemm_tile2(const unsigned short* __restrict__ A0bf,
                                  const unsigned short* __restrict__ A1bf, int lda,
                                  int arow, const float* __restrict__ brow, int lane,
                                  v8f& c0, v8f& c1)
{
    const int aoff = (lane & 16) ? 16 : 0;
    const int koff = (lane & 16) ? 16 : 0;
    const unsigned short* ap0 = A0bf + (size_t)arow * lda + aoff;
    const unsigned short* ap1 = A1bf + (size_t)arow * lda + aoff;
#pragma unroll 4
    for (int kb = 0; kb < K; kb += 32) {
        const float* bp = brow + kb + koff;
        __builtin_prefetch(bp + 128, 0, 1);
        v16bf a0 = load_a16(ap0 + kb);
        v16bf a1 = load_a16(ap1 + kb);
        v16bf b  = load_b16(bp);
        c0 = __builtin_amdgcn_wmma_f32_16x16x32_bf16(false, a0, false, b,
                                                     (short)0, c0, false, false);
        c1 = __builtin_amdgcn_wmma_f32_16x16x32_bf16(false, a1, false, b,
                                                     (short)0, c1, false, false);
    }
}

// ---------------------------------------------------------------------------
// Prep: row norms of new_embeds, pre-convert the three 128-row A matrices to
// bf16 in workspace (L2-resident thereafter).
// ---------------------------------------------------------------------------
__global__ void __launch_bounds__(256) cll_prep_kernel(
    const float* __restrict__ olde, const float* __restrict__ newe,
    const float* __restrict__ nl,
    unsigned short* __restrict__ oldeB, unsigned short* __restrict__ neweB,
    unsigned short* __restrict__ nlB)
{
    const int row = blockIdx.x;
    const int t   = threadIdx.x;
    __shared__ float red[256];
    float ss = 0.f;
    for (int d = t; d < Dn; d += 256) {
        float v = newe[(size_t)row * Dn + d];
        ss += v * v;
    }
    red[t] = ss;
    __syncthreads();
    for (int off = 128; off > 0; off >>= 1) {
        if (t < off) red[t] += red[t + off];
        __syncthreads();
    }
    const float inv = 1.0f / fmaxf(sqrtf(red[0]), 1e-12f);
    __bf16* oe = reinterpret_cast<__bf16*>(oldeB);
    __bf16* ne = reinterpret_cast<__bf16*>(neweB);
    __bf16* nb = reinterpret_cast<__bf16*>(nlB);
    for (int d = t; d < Dn; d += 256) {
        ne[(size_t)row * Dn + d] = (__bf16)(newe[(size_t)row * Dn + d] * inv);
        oe[(size_t)row * Dn + d] = (__bf16)olde[(size_t)row * Dn + d];
    }
    for (int d = t; d < Cn; d += 256)
        nb[(size_t)row * Cn + d] = (__bf16)nl[(size_t)row * Cn + d];
}

// ---------------------------------------------------------------------------
// Main fused kernel: one workgroup (8 waves = all 128 rows) per 16-queue-row
// tile.  3 WMMA GEMMs + mask/weight + online-softmax partial stats.
// partials[qb][row][8] = {m1, Z1, Smw_s1, m2, Z2, Smw_s2, Smw, Sm}
// ---------------------------------------------------------------------------
__global__ void __launch_bounds__(256, 1) cll_tile_kernel(
    const float* __restrict__ olde, const float* __restrict__ oldl,
    const float* __restrict__ featq, const float* __restrict__ logitq,
    const int* __restrict__ labels, const int* __restrict__ qlabels,
    const int* __restrict__ headerp,
    const unsigned short* __restrict__ oldeB,
    const unsigned short* __restrict__ neweB,
    const unsigned short* __restrict__ nlB,
    float* __restrict__ partials)
{
    const int lane = threadIdx.x & 31;
    const int wv   = threadIdx.x >> 5;       // 0..7 -> A-row block
    const int qb   = blockIdx.x;             // 0..NB-1
    const int col  = lane & 15;              // B column within tile
    const int q    = qb * 16 + col;          // queue row this lane feeds

    int hdr = headerp[0] % Qn; if (hdr < 0) hdr += Qn;
    int j = q - hdr; if (j < 0) j += Qn;
    const bool ovl = (j < Nn);               // virtual circular-queue update
    const float* frow = ovl ? (olde + (size_t)j * Dn) : (featq  + (size_t)q * Dn);
    const float* lrow = ovl ? (oldl + (size_t)j * Cn) : (logitq + (size_t)q * Cn);
    const int qlab = ovl ? labels[j] : qlabels[q];

    const int mbase = wv * 16;
    const int arow  = mbase + col;

    v8f cw = {}, cs1 = {};
    gemm_tile2<Dn>(oldeB, neweB, Dn, arow, frow, lane, cw, cs1); // shared feat B
    v8f cs2 = gemm_tile<Cn>(nlB, Cn, arow, lrow, lane);          // new_logits @ logitᵀ

    const int rbase = mbase + ((lane & 16) ? 8 : 0);        // C-layout row base
    float st[8][8];
#pragma unroll
    for (int r = 0; r < 8; ++r) {
        const int   row = rbase + r;
        const float wgt = 0.5f * (cw[r] + 1.0f);
        const float mk  = (labels[row] == qlab) ? 1.0f : 0.0f;
        const float mw  = mk * wgt;
        const float s1  = cs1[r];     // TEMP == 1.0
        const float s2  = cs2[r];
        st[r][0] = s1;  st[r][1] = 1.0f; st[r][2] = mw * s1;
        st[r][3] = s2;  st[r][4] = 1.0f; st[r][5] = mw * s2;
        st[r][6] = mw;  st[r][7] = mk;
    }
    // butterfly-merge the 16 columns (lanes within each 16-lane half)
#pragma unroll
    for (int m = 1; m <= 8; m <<= 1) {
#pragma unroll
        for (int r = 0; r < 8; ++r) {
            float om1 = __shfl_xor(st[r][0], m);
            float oz1 = __shfl_xor(st[r][1], m);
            float os1 = __shfl_xor(st[r][2], m);
            float om2 = __shfl_xor(st[r][3], m);
            float oz2 = __shfl_xor(st[r][4], m);
            float os2 = __shfl_xor(st[r][5], m);
            float osw = __shfl_xor(st[r][6], m);
            float osm = __shfl_xor(st[r][7], m);
            float nm1 = fmaxf(st[r][0], om1);
            st[r][1] = st[r][1] * __expf(st[r][0] - nm1) + oz1 * __expf(om1 - nm1);
            st[r][0] = nm1;
            float nm2 = fmaxf(st[r][3], om2);
            st[r][4] = st[r][4] * __expf(st[r][3] - nm2) + oz2 * __expf(om2 - nm2);
            st[r][3] = nm2;
            st[r][2] += os1; st[r][5] += os2; st[r][6] += osw; st[r][7] += osm;
        }
    }
    if (col == 0) {
#pragma unroll
        for (int r = 0; r < 8; ++r) {
            float* p = partials + ((size_t)qb * Nn + (rbase + r)) * 8;
#pragma unroll
            for (int k = 0; k < 8; ++k) p[k] = st[r][k];
        }
    }
}

// ---------------------------------------------------------------------------
// Reduction stage 1: merge the NB per-tile partials for each row.
// ---------------------------------------------------------------------------
__device__ inline void merge8(float* a, const float* b)
{
    float nm = fmaxf(a[0], b[0]);
    a[1] = a[1] * __expf(a[0] - nm) + b[1] * __expf(b[0] - nm);
    a[0] = nm;
    nm = fmaxf(a[3], b[3]);
    a[4] = a[4] * __expf(a[3] - nm) + b[4] * __expf(b[3] - nm);
    a[3] = nm;
    a[2] += b[2]; a[5] += b[5]; a[6] += b[6]; a[7] += b[7];
}

__global__ void __launch_bounds__(256) cll_rowreduce_kernel(
    const float* __restrict__ partials, float* __restrict__ rowvals)
{
    const int row = blockIdx.x;
    const int t   = threadIdx.x;
    float acc[8] = {-3.0e38f, 0.f, 0.f, -3.0e38f, 0.f, 0.f, 0.f, 0.f};
    for (int b = t; b < NB; b += 256) {
        const float* p = partials + ((size_t)b * Nn + row) * 8;
        float v[8];
#pragma unroll
        for (int k = 0; k < 8; ++k) v[k] = p[k];
        merge8(acc, v);
    }
    __shared__ float sh[256][8];
#pragma unroll
    for (int k = 0; k < 8; ++k) sh[t][k] = acc[k];
    __syncthreads();
    for (int off = 128; off > 0; off >>= 1) {
        if (t < off) merge8(sh[t], sh[t + off]);
        __syncthreads();
    }
    if (t == 0) {
        float lse1 = sh[0][0] + __logf(sh[0][1]);     // rowmax + log(sum exp)
        float lse2 = sh[0][3] + __logf(sh[0][4]);
        float inv  = 1.0f / sh[0][7];                 // / sum(mask)
        rowvals[row]      = (sh[0][2] - lse1 * sh[0][6]) * inv;
        rowvals[Nn + row] = (sh[0][5] - lse2 * sh[0][6]) * inv;
    }
}

// Reduction stage 2: mean over rows, negate -> two scalar losses.
__global__ void __launch_bounds__(128) cll_final_kernel(
    const float* __restrict__ rowvals, float* __restrict__ out)
{
    const int t = threadIdx.x;
    __shared__ float s1[128], s2[128];
    s1[t] = rowvals[t];
    s2[t] = rowvals[Nn + t];
    __syncthreads();
    for (int off = 64; off > 0; off >>= 1) {
        if (t < off) { s1[t] += s1[t + off]; s2[t] += s2[t + off]; }
        __syncthreads();
    }
    if (t == 0) {
        out[0] = -s1[0] / (float)Nn;
        out[1] = -s2[0] / (float)Nn;
    }
}

// ---------------------------------------------------------------------------
extern "C" void kernel_launch(void* const* d_in, const int* in_sizes, int n_in,
                              void* d_out, int out_size, void* d_ws, size_t ws_size,
                              hipStream_t stream)
{
    const float* olde    = (const float*)d_in[0];
    const float* oldl    = (const float*)d_in[1];
    const float* newe    = (const float*)d_in[2];
    const float* nl      = (const float*)d_in[3];
    const int*   labels  = (const int*)d_in[4];
    const float* featq   = (const float*)d_in[5];
    const float* logitq  = (const float*)d_in[6];
    const int*   qlabels = (const int*)d_in[7];
    const int*   header  = (const int*)d_in[8];
    float*       out     = (float*)d_out;

    char* ws = (char*)d_ws;
    unsigned short* oldeB = (unsigned short*)ws;                 // 128*512 bf16
    unsigned short* neweB = oldeB + (size_t)Nn * Dn;             // 128*512 bf16
    unsigned short* nlB   = neweB + (size_t)Nn * Dn;             // 128*8192 bf16
    size_t bfBytes  = (size_t)2 * Nn * Dn * 2 + (size_t)Nn * Cn * 2;  // 2,359,296
    float* partials = (float*)(ws + bfBytes);                    // NB*128*8 f32
    float* rowvals  = partials + (size_t)NB * Nn * 8;            // 256 f32

    cll_prep_kernel<<<Nn, 256, 0, stream>>>(olde, newe, nl, oldeB, neweB, nlB);
    cll_tile_kernel<<<NB, 256, 0, stream>>>(olde, oldl, featq, logitq,
                                            labels, qlabels, header,
                                            oldeB, neweB, nlB, partials);
    cll_rowreduce_kernel<<<Nn, 256, 0, stream>>>(partials, rowvals);
    cll_final_kernel<<<1, 128, 0, stream>>>(rowvals, out);
}
